// SimpleGRU_3977139716147
// MI455X (gfx1250) — matrix operand order repaired and checked by reading, not compiled, not run on hardware
//
#include <hip/hip_runtime.h>

// ---------------------------------------------------------------------------
// SimpleGRU for MI455X (gfx1250, wave32, WMMA).
//
// Cost model (MI455X, 23.3 TB/s HBM):
//   * FC GEMM [4096,100]@[100,32000] -> 524MB fp32 out, 26.2 GFLOP.
//     BW floor ~23us; needs >1.1 PFLOP/s => v_wmma_f32_16x16x32_bf16.
//   * x@W_x* projections: parallel over all (t,b) -> precompute kernel.
//   * h-recurrence: strictly sequential, 240K MAC/step. Single workgroup,
//     W_h* pinned in VGPRs as WMMA B operands, h in LDS as bf16 A tile.
//     Critical path per step: 8 ds_load_b128 -> 2 parallel WMMA chains ->
//     v_rcp-based sigmoid/tanh (no IEEE div) -> LDS/global b16 scatter.
// ---------------------------------------------------------------------------

typedef __bf16 bf16_t;
typedef __attribute__((ext_vector_type(16))) __bf16 v16bf;
typedef __attribute__((ext_vector_type(8)))  __bf16 v8bf;
typedef __attribute__((ext_vector_type(8)))  float  v8f;

#define GB   8
#define GT   512
#define GE   50
#define GH   100
#define GV   32000
#define GM   (GB * GT)      // 4096 GEMM rows, m = b*512 + t
#define GK   128            // K padded 100 -> 128 (4 chunks of 32)
#define MT   (GM / 16)      // 256 M tiles
#define NTB  (GV / 16)      // 2000 N tiles
#define NW   21             // recurrence waves: 7 r-tiles + 7 z-tiles + 7 h-tiles

// workspace layout (bytes); total required = 20,250,624 B
static constexpr size_t XOFF = 0;                         // Xpack: 512*21*32*8 f32 = 11,010,048
static constexpr size_t AOFF = 11010048;                  // Apack: 256*4*32*32 B  =  1,048,576
static constexpr size_t BOFF = AOFF + 1048576;            // Bpack: 2000*4*32*32 B =  8,192,000

__device__ __forceinline__ v8f wmma_bf16(v16bf a, v16bf b, v8f c) {
  return __builtin_amdgcn_wmma_f32_16x16x32_bf16(false, a, false, b, (short)0, c,
                                                 false, false);
}

// Fast transcendentals on the serial critical path: v_rcp_f32 instead of the
// correctly-rounded division sequence (div_scale/div_fmas/div_fixup).
__device__ __forceinline__ float fast_sigmoid(float x) {
  return __builtin_amdgcn_rcpf(1.f + __expf(-x));
}
__device__ __forceinline__ float fast_tanh(float x) {
  return 1.f - 2.f * __builtin_amdgcn_rcpf(__expf(2.f * x) + 1.f);
}

// A-operand placement (16-bit A 16x32 per ISA 7.12.2) for row r, k-in-chunk kk.
__device__ __forceinline__ void a_slot(int r, int kk, int& lane, int& e) {
  if (kk < 8)       { lane = r;      e = kk;      }
  else if (kk < 16) { lane = r + 16; e = kk - 8;  }
  else if (kk < 24) { lane = r;      e = kk - 8;  }
  else              { lane = r + 16; e = kk - 16; }
}

// ---------------------------------------------------------------------------
// K0: zero Apack (covers the K=100..127 / row padding once per call)
// ---------------------------------------------------------------------------
__global__ void zero_apack(unsigned int* p) {
  int i = blockIdx.x * 256 + threadIdx.x;
  if (i < (int)(1048576 / 4)) p[i] = 0u;
}

// ---------------------------------------------------------------------------
// K1: pack fc_W (fp32 [100][32000]) into bf16 WMMA B-operand layout.
//     Bpack[((tile*4+chunk)*32 + lane)*16 + e]
// ---------------------------------------------------------------------------
__global__ void pack_b(const float* __restrict__ fcW, bf16_t* __restrict__ Bp) {
  int idx = blockIdx.x * 256 + threadIdx.x;
  if (idx >= GK * GV) return;
  int n = idx % GV;
  int k = idx / GV;
  float v = (k < GH) ? fcW[k * GV + n] : 0.f;
  int tile = n >> 4, colb = n & 15, c = k >> 5, kk = k & 31;
  int lanep = colb + ((kk >> 3) & 1) * 16;
  int e = (kk & 7) + (kk >> 4) * 8;
  Bp[(((size_t)tile * 4 + c) * 32 + lanep) * 16 + e] = (bf16_t)v;
}

// ---------------------------------------------------------------------------
// K2: precompute X = embed(x) @ W_x{r,z,h} + b_{r,z,h} for every (t,b),
//     written directly in WMMA C-tile layout: Xpack[((t*21+w)*32+lane)*8+v]
//     wave w<7: r-gate tile w; w<14: z tile w-7; w>=14: h tile w-14.
// ---------------------------------------------------------------------------
__global__ void precompute_x(const int* __restrict__ tokens,
                             const float* __restrict__ emb,
                             const float* __restrict__ Wxr, const float* __restrict__ br,
                             const float* __restrict__ Wxz, const float* __restrict__ bz,
                             const float* __restrict__ Wxh, const float* __restrict__ bh,
                             float* __restrict__ Xpack) {
  int idx = blockIdx.x * 256 + threadIdx.x;
  if (idx >= GT * NW * 32 * 8) return;
  int v    = idx & 7;
  int lane = (idx >> 3) & 31;
  int w    = (idx >> 8) % NW;
  int t    = idx / (NW * 256);
  float val = 0.f;
  if (lane < 16) {                       // C-tile: only lanes 0-15 carry b=v<8
    const float* Wx; const float* bb; int nt;
    if (w < 7)       { Wx = Wxr; bb = br; nt = w; }
    else if (w < 14) { Wx = Wxz; bb = bz; nt = w - 7; }
    else             { Wx = Wxh; bb = bh; nt = w - 14; }
    int j = nt * 16 + (lane & 15);
    if (j < GH) {
      int b = v;
      int tok = tokens[b * GT + t];
      const float* er = emb + (size_t)tok * GE;
      float s = bb[j];
#pragma unroll
      for (int e = 0; e < GE; ++e) s += er[e] * Wx[e * GH + j];
      val = s;
    }
  }
  Xpack[idx] = val;
}

// ---------------------------------------------------------------------------
// K3: sequential GRU recurrence. One workgroup, 21 waves.
//     W_hr/W_hz/W_hh live in VGPRs as WMMA B operands for the whole t-loop.
//     h lives in LDS as a bf16 16x128 A tile (rows>=8 & cols>=100 are zero).
//     Emits h states straight into Apack (WMMA A-operand layout) for K4.
// ---------------------------------------------------------------------------
__global__ __launch_bounds__(NW * 32) void gru_seq(const float* __restrict__ Whr,
                                                   const float* __restrict__ Whz,
                                                   const float* __restrict__ Whh,
                                                   char* __restrict__ ws) {
  __shared__ __align__(16) bf16_t h_bf[16][GK];
  __shared__ __align__(16) bf16_t rh_bf[16][GK];
  __shared__ float h_f[8][112];
  __shared__ float z_f[8][112];

  const int tid  = threadIdx.x;
  const int w    = tid >> 5;
  const int lane = tid & 31;
  const int col  = lane & 15;
  const bool lo  = lane < 16;
  const int abase = lo ? 0 : 8;          // A-operand K base per half-wave

  for (int i = tid; i < 16 * GK; i += NW * 32) {
    (&h_bf[0][0])[i]  = (bf16_t)0.f;
    (&rh_bf[0][0])[i] = (bf16_t)0.f;
  }
  for (int i = tid; i < 8 * 112; i += NW * 32) {
    (&h_f[0][0])[i] = 0.f;
    (&z_f[0][0])[i] = 0.f;
  }

  const float* Wg = (w < 7) ? Whr : (w < 14) ? Whz : Whh;
  const int nt    = (w < 7) ? w : (w < 14) ? (w - 7) : (w - 14);

  // Pin this wave's 16-column slice of W_h* as 4 WMMA B operands (once).
  v16bf bw[4];
#pragma unroll
  for (int c = 0; c < 4; ++c) {
#pragma unroll
    for (int i = 0; i < 16; ++i) {
      int k = c * 32 + (i & 8) * 2 + (lo ? 0 : 8) + (i & 7);
      int j = nt * 16 + col;
      float wv = (k < GH && j < GH) ? Wg[k * GH + j] : 0.f;
      bw[c][i] = (bf16_t)wv;
    }
  }
  __syncthreads();

  const float* Xpack = (const float*)(ws + XOFF);
  bf16_t*      Apack = (bf16_t*)(ws + AOFF);

  for (int t = 0; t < GT; ++t) {
    const float4* xp =
        (const float4*)(Xpack + (((size_t)t * NW + w) * 32 + lane) * 8);
    if (t + 1 < GT)
      __builtin_prefetch(xp + (size_t)NW * 64, 0, 0);   // next step's X row

    if (w < 14) {                        // ---- phase A: r and z gates ----
      float4 x0 = xp[0], x1 = xp[1];
      // Preload all 4 A chunks (8 back-to-back ds_load_b128), single wait.
      v16bf a[4];
#pragma unroll
      for (int c = 0; c < 4; ++c) {
        v8bf alo = *(const v8bf*)&h_bf[col][c * 32 + abase];
        v8bf ahi = *(const v8bf*)&h_bf[col][c * 32 + 16 + abase];
#pragma unroll
        for (int i = 0; i < 8; ++i) { a[c][i] = alo[i]; a[c][8 + i] = ahi[i]; }
      }
      // Two independent accumulation chains -> half the dependent WMMA chain.
      v8f acc0 = {x0.x, x0.y, x0.z, x0.w, x1.x, x1.y, x1.z, x1.w};
      v8f acc1 = {};
      acc0 = wmma_bf16(a[0], bw[0], acc0);
      acc1 = wmma_bf16(a[1], bw[1], acc1);
      acc0 = wmma_bf16(a[2], bw[2], acc0);
      acc1 = wmma_bf16(a[3], bw[3], acc1);
      v8f acc = acc0 + acc1;
      float s[8];
#pragma unroll
      for (int v = 0; v < 8; ++v) s[v] = fast_sigmoid(acc[v]);
      if (lo) {
        int j = nt * 16 + col;
        if (j < GH) {
          if (w < 7) {
#pragma unroll
            for (int v = 0; v < 8; ++v)
              rh_bf[v][j] = (bf16_t)(s[v] * h_f[v][j]);
          } else {
#pragma unroll
            for (int v = 0; v < 8; ++v) z_f[v][j] = s[v];
          }
        }
      }
    }
    __syncthreads();

    if (w >= 14) {                       // ---- phase B: h~ and h update ----
      float4 x0 = xp[0], x1 = xp[1];
      v16bf a[4];
#pragma unroll
      for (int c = 0; c < 4; ++c) {
        v8bf alo = *(const v8bf*)&rh_bf[col][c * 32 + abase];
        v8bf ahi = *(const v8bf*)&rh_bf[col][c * 32 + 16 + abase];
#pragma unroll
        for (int i = 0; i < 8; ++i) { a[c][i] = alo[i]; a[c][8 + i] = ahi[i]; }
      }
      v8f acc0 = {x0.x, x0.y, x0.z, x0.w, x1.x, x1.y, x1.z, x1.w};
      v8f acc1 = {};
      acc0 = wmma_bf16(a[0], bw[0], acc0);
      acc1 = wmma_bf16(a[1], bw[1], acc1);
      acc0 = wmma_bf16(a[2], bw[2], acc0);
      acc1 = wmma_bf16(a[3], bw[3], acc1);
      v8f acc = acc0 + acc1;
      float ht[8];
#pragma unroll
      for (int v = 0; v < 8; ++v) ht[v] = fast_tanh(acc[v]);
      if (lo) {
        int j = nt * 16 + col;
        if (j < GH) {
#pragma unroll
          for (int v = 0; v < 8; ++v) {
            float z  = z_f[v][j];
            float hn = z * ht[v] + (1.f - z) * h_f[v][j];
            h_f[v][j]  = hn;
            h_bf[v][j] = (bf16_t)hn;
            // emit into Apack (A-operand layout) for the FC GEMM
            int m = v * GT + t;          // row ordering matches out[B,T,V]
            int mt = m >> 4, r = m & 15;
            int c = j >> 5, kk = j & 31;
            int lp, e;
            a_slot(r, kk, lp, e);
            Apack[(((size_t)mt * 4 + c) * 32 + lp) * 16 + e] = (bf16_t)hn;
          }
        }
      }
    }
    __syncthreads();
  }
}

// ---------------------------------------------------------------------------
// K4: out[m][n] = Apack @ Bpack + fc_b.  Per wave: 4 N-tiles (64 cols) with
//     B pinned in 128 VGPRs across a 16-M-tile loop; 16 wmma per M tile.
//     A chunks preloaded (8 global_load_b128, one wait) before the WMMAs;
//     the 4 N-tile accumulator chains are independent and interleave.
//     grid = (32000/256, 256/16), block = 128 (4 waves).
// ---------------------------------------------------------------------------
__global__ __launch_bounds__(128) void fc_gemm(const char* __restrict__ ws,
                                               const float* __restrict__ fcb,
                                               float* __restrict__ out) {
  const int lane = threadIdx.x & 31;
  const int wv   = threadIdx.x >> 5;
  const int col  = lane & 15;
  const bool lo  = lane < 16;
  const int nt0  = (blockIdx.x * 4 + wv) * 4;  // first of 4 N tiles

  const bf16_t* Bp = (const bf16_t*)(ws + BOFF);
  const bf16_t* Ap = (const bf16_t*)(ws + AOFF);

  v16bf breg[4][4];                            // [ntile][kchunk], pinned
#pragma unroll
  for (int i = 0; i < 4; ++i) {
#pragma unroll
    for (int c = 0; c < 4; ++c) {
      const v8bf* p =
          (const v8bf*)(Bp + ((((size_t)(nt0 + i)) * 4 + c) * 32 + lane) * 16);
      v8bf blo = p[0], bhi = p[1];
#pragma unroll
      for (int e = 0; e < 8; ++e) { breg[i][c][e] = blo[e]; breg[i][c][8 + e] = bhi[e]; }
    }
  }
  float bias[4];
#pragma unroll
  for (int i = 0; i < 4; ++i) bias[i] = fcb[(nt0 + i) * 16 + col];

  const int mt0 = blockIdx.y * 16;
  for (int mtl = 0; mtl < 16; ++mtl) {
    const int mt = mt0 + mtl;
    if (mtl + 1 < 16)
      __builtin_prefetch(Ap + (((size_t)(mt + 1) * 4) * 32 + lane) * 16, 0, 0);

    // Preload all 4 A chunks, then issue 16 WMMAs (4 independent chains).
    v16bf a[4];
#pragma unroll
    for (int c = 0; c < 4; ++c) {
      const v8bf* pa = (const v8bf*)(Ap + (((size_t)mt * 4 + c) * 32 + lane) * 16);
      v8bf alo = pa[0], ahi = pa[1];
#pragma unroll
      for (int e = 0; e < 8; ++e) { a[c][e] = alo[e]; a[c][8 + e] = ahi[e]; }
    }
    v8f acc[4] = {};
#pragma unroll
    for (int c = 0; c < 4; ++c)
#pragma unroll
      for (int i = 0; i < 4; ++i) acc[i] = wmma_bf16(a[c], breg[i][c], acc[i]);

    const int rowbase = mt * 16 + (lo ? 0 : 8);
#pragma unroll
    for (int i = 0; i < 4; ++i) {
      const int n = (nt0 + i) * 16 + col;
#pragma unroll
      for (int v = 0; v < 8; ++v)
        out[(size_t)(rowbase + v) * GV + n] = acc[i][v] + bias[i];
    }
  }
}

// ---------------------------------------------------------------------------
extern "C" void kernel_launch(void* const* d_in, const int* in_sizes, int n_in,
                              void* d_out, int out_size, void* d_ws, size_t ws_size,
                              hipStream_t stream) {
  const int*   tokens = (const int*)d_in[0];
  const float* emb    = (const float*)d_in[1];
  const float* Whr    = (const float*)d_in[2];
  const float* Wxr    = (const float*)d_in[3];
  const float* br     = (const float*)d_in[4];
  const float* Whz    = (const float*)d_in[5];
  const float* Wxz    = (const float*)d_in[6];
  const float* bz     = (const float*)d_in[7];
  const float* Whh    = (const float*)d_in[8];
  const float* Wxh    = (const float*)d_in[9];
  const float* bh     = (const float*)d_in[10];
  const float* fcW    = (const float*)d_in[11];
  const float* fcb    = (const float*)d_in[12];
  float* out = (float*)d_out;
  char*  ws  = (char*)d_ws;

  zero_apack<<<(1048576 / 4 + 255) / 256, 256, 0, stream>>>(
      (unsigned int*)(ws + AOFF));
  pack_b<<<(GK * GV + 255) / 256, 256, 0, stream>>>(fcW, (bf16_t*)(ws + BOFF));
  precompute_x<<<(GT * NW * 256 + 255) / 256, 256, 0, stream>>>(
      tokens, emb, Wxr, br, Wxz, bz, Wxh, bh, (float*)(ws + XOFF));
  gru_seq<<<1, NW * 32, 0, stream>>>(Whr, Whz, Whh, ws);
  fc_gemm<<<dim3(GV / 256, MT / 16), 128, 0, stream>>>(ws, fcb, out);
}
